// DeepRGCN_14834817040645
// MI455X (gfx1250) — matrix operand order repaired and checked by reading, not compiled, not run on hardware
//
#include <hip/hip_runtime.h>

// Problem constants (from reference)
#define NN 50000
#define RR 64
#define DD 120
#define NBLK 4
#define BDIM 30
#define LL 2
#define EE 640000
#define TILE_E 64                       // edges per workgroup (4 waves x 16)
#define MAX_TILES (EE / TILE_E + RR)    // 10064 upper bound

typedef __attribute__((ext_vector_type(16))) __bf16 bf16x16;
typedef __attribute__((ext_vector_type(8)))  float  f32x8;

union BFrag { bf16x16 v; unsigned short u[16]; unsigned u32[8]; };

__device__ __forceinline__ unsigned short f2bf(float f) {
  unsigned u = __float_as_uint(f);
  u += 0x7FFFu + ((u >> 16) & 1u);              // round-to-nearest-even
  return (unsigned short)(u >> 16);
}
__device__ __forceinline__ float bf2f(unsigned short h) {
  return __uint_as_float(((unsigned)h) << 16);
}

// ---------------- utility kernels ----------------
__global__ void k_zero_u32(unsigned* __restrict__ p, long n) {
  long i = (long)blockIdx.x * blockDim.x + threadIdx.x;
  if (i < n) p[i] = 0u;
}
__global__ void k_zero_f32(float* __restrict__ p, long n) {
  long i = (long)blockIdx.x * blockDim.x + threadIdx.x;
  if (i < n) p[i] = 0.0f;
}

__global__ void k_count(const int* __restrict__ dst, const int* __restrict__ et,
                        unsigned* __restrict__ cnt, unsigned* __restrict__ rel_cnt) {
  int e = blockIdx.x * blockDim.x + threadIdx.x;
  if (e >= EE) return;
  atomicAdd(&cnt[(size_t)dst[e] * RR + et[e]], 1u);
  atomicAdd(&rel_cnt[et[e]], 1u);
}

__global__ void k_scan_tiles(const unsigned* __restrict__ rel_cnt,
                             unsigned* __restrict__ rel_off,
                             unsigned* __restrict__ rel_cur,
                             uint2* __restrict__ tiles,
                             unsigned* __restrict__ n_tiles) {
  if (threadIdx.x != 0 || blockIdx.x != 0) return;
  unsigned off = 0;
  for (int r = 0; r < RR; ++r) { rel_off[r] = off; rel_cur[r] = off; off += rel_cnt[r]; }
  rel_off[RR] = off;
  unsigned t = 0;
  for (int r = 0; r < RR; ++r)
    for (unsigned s = rel_off[r]; s < rel_off[r + 1]; s += TILE_E) {
      tiles[t].x = s; tiles[t].y = (unsigned)r; ++t;
    }
  *n_tiles = t;
}

__global__ void k_scatter(const int* __restrict__ et, unsigned* __restrict__ rel_cur,
                          unsigned* __restrict__ eord) {
  int e = blockIdx.x * blockDim.x + threadIdx.x;
  if (e >= EE) return;
  unsigned pos = atomicAdd(&rel_cur[et[e]], 1u);
  eord[pos] = (unsigned)e;
}

// split fp32 rel_W into bf16 hi/lo
__global__ void k_convW(const float* __restrict__ w, unsigned short* __restrict__ wh,
                        unsigned short* __restrict__ wl, long n) {
  long i = (long)blockIdx.x * blockDim.x + threadIdx.x;
  if (i >= n) return;
  float f = w[i];
  unsigned short h = f2bf(f);
  wh[i] = h;
  wl[i] = f2bf(f - bf2f(h));
}

// split activations into bf16 hi/lo, rows padded 120 -> 128 (zeros)
__global__ void k_convX(const float* __restrict__ x, unsigned short* __restrict__ xh,
                        unsigned short* __restrict__ xl) {
  long i = (long)blockIdx.x * blockDim.x + threadIdx.x;
  if (i >= (long)NN * 128) return;
  int r = (int)(i >> 7), c = (int)(i & 127);
  float f = (c < DD) ? x[(size_t)r * DD + c] : 0.0f;
  unsigned short h = f2bf(f);
  xh[i] = h;
  xl[i] = f2bf(f - bf2f(h));
}

// root W: transpose to [l][n(128)][k(128)], zero-padded, bf16 hi/lo
__global__ void k_convRW(const float* __restrict__ w, unsigned short* __restrict__ wh,
                         unsigned short* __restrict__ wl) {
  long i = (long)blockIdx.x * blockDim.x + threadIdx.x;
  if (i >= (long)LL * 128 * 128) return;
  int l = (int)(i >> 14);
  int rem = (int)(i & 16383);
  int n = rem >> 7, k = rem & 127;
  float f = (n < DD && k < DD) ? w[(size_t)l * DD * DD + (size_t)k * DD + n] : 0.0f;
  unsigned short h = f2bf(f);
  wh[i] = h;
  wl[i] = f2bf(f - bf2f(h));
}

// ---------------- WMMA message kernel ----------------
// One workgroup = 64 edges of ONE relation (4 waves x 16 edges).
// Branch-free fragment builds from zero-padded bf16 LDS; results land in an LDS
// message tile (union with X staging), then row-uniform coalesced atomic scatter.
__global__ __launch_bounds__(128)
void k_msg(const unsigned short* __restrict__ Xh, const unsigned short* __restrict__ Xl,
           const int* __restrict__ src, const int* __restrict__ dst,
           const unsigned* __restrict__ eord,
           const unsigned* __restrict__ cnt,
           const unsigned* __restrict__ rel_off,
           const uint2* __restrict__ tiles,
           const unsigned* __restrict__ n_tiles,
           const unsigned short* __restrict__ Wh,
           const unsigned short* __restrict__ Wl,
           float* __restrict__ agg) {
  union ShmU {
    unsigned short xh[4][2][16][128];   // [wave][hi/lo][row][col] : 32768 B
    float msg[4][16][128];              // same bytes, reused after frags built
  };
  __shared__ __align__(16) ShmU su;
  __shared__ unsigned short sWh[NBLK][32][32];   // [block][col][k] transposed, padded
  __shared__ unsigned short sWl[NBLK][32][32];
  __shared__ int   sSrc[4][16];
  __shared__ int   sDst[4][16];
  __shared__ float sNrm[4][16];

  unsigned bid = blockIdx.x;
  if (bid >= *n_tiles) return;
  uint2 tl = tiles[bid];
  unsigned start = tl.x, rel = tl.y;
  unsigned segEnd = rel_off[rel + 1];

  int tid = threadIdx.x;
  int wid = tid >> 5;
  int lane = tid & 31;

  // zero W pads (u32 granularity), then fill transposed
  for (int i = tid; i < NBLK * 32 * 32 / 2; i += 128) {
    reinterpret_cast<unsigned*>(&sWh[0][0][0])[i] = 0u;
    reinterpret_cast<unsigned*>(&sWl[0][0][0])[i] = 0u;
  }
  __syncthreads();
  for (int i = tid; i < NBLK * BDIM * BDIM; i += 128) {
    int b = i / (BDIM * BDIM);
    int r2 = i - b * (BDIM * BDIM);
    int k = r2 / BDIM;
    int c = r2 - k * BDIM;
    size_t g = (size_t)rel * NBLK * BDIM * BDIM + i;
    sWh[b][c][k] = Wh[g];
    sWl[b][c][k] = Wl[g];
  }
  if (lane < 16) {
    unsigned eidx = start + (unsigned)wid * 16u + (unsigned)lane;
    if (eidx < segEnd) {
      unsigned e = eord[eidx];
      int d = dst[e];
      unsigned c = cnt[(size_t)d * RR + rel];
      sSrc[wid][lane] = src[e];
      sDst[wid][lane] = d;
      sNrm[wid][lane] = 1.0f / (float)(c > 0u ? c : 1u);
    } else {
      sSrc[wid][lane] = -1;
      sDst[wid][lane] = -1;
      sNrm[wid][lane] = 0.0f;
    }
  }
  __syncthreads();

  // stage 16 bf16 hi/lo rows per wave (uint4 = 16B chunks; zero for pad edges)
  for (int i = lane; i < 2 * 16 * 16; i += 32) {        // 512 uint4
    int part = i >> 8;
    int rem = i & 255;
    int r = rem >> 4;
    int q = rem & 15;
    int s = sSrc[wid][r];
    uint4 val = make_uint4(0u, 0u, 0u, 0u);
    const unsigned short* Xsrc = part ? Xl : Xh;
    if (s >= 0) val = *reinterpret_cast<const uint4*>(Xsrc + (size_t)s * 128 + q * 8);
    *reinterpret_cast<uint4*>(&su.xh[wid][part][r][q * 8]) = val;
  }
  __syncthreads();

  int m = lane & 15;
  int half = lane >> 4;

  // build ALL A fragments first (frees the xh region for msg reuse)
  BFrag ah[NBLK], al[NBLK];
  #pragma unroll
  for (int b = 0; b < NBLK; ++b) {
    const unsigned short* xrh = &su.xh[wid][0][m][0];
    const unsigned short* xrl = &su.xh[wid][1][m][0];
    #pragma unroll
    for (int v = 0; v < 8; ++v) {
      int kb = half * 8 + ((v < 4) ? 2 * v : 16 + 2 * (v - 4));
      int c0 = b * BDIM + kb;            // even; pads (>=120) are zero
      ah[b].u32[v] = *reinterpret_cast<const unsigned*>(xrh + c0);
      al[b].u32[v] = *reinterpret_cast<const unsigned*>(xrl + c0);
    }
  }

  #pragma unroll
  for (int b = 0; b < NBLK; ++b) {
    #pragma unroll
    for (int nt = 0; nt < 2; ++nt) {
      int colL = nt * 16 + m;            // 0..31, padded cols are zero
      BFrag bh, bl;
      #pragma unroll
      for (int v = 0; v < 8; ++v) {
        int kk = half * 8 + ((v < 4) ? 2 * v : 16 + 2 * (v - 4));      // even; pads (>=30) are zero
        bh.u32[v] = *reinterpret_cast<const unsigned*>(&sWh[b][colL][kk]);
        bl.u32[v] = *reinterpret_cast<const unsigned*>(&sWl[b][colL][kk]);
      }
      f32x8 acc = {};
      acc = __builtin_amdgcn_wmma_f32_16x16x32_bf16(false, ah[b].v, false, bh.v, (short)0, acc, false, false);
      acc = __builtin_amdgcn_wmma_f32_16x16x32_bf16(false, ah[b].v, false, bl.v, (short)0, acc, false, false);
      acc = __builtin_amdgcn_wmma_f32_16x16x32_bf16(false, al[b].v, false, bh.v, (short)0, acc, false, false);

      int colInBlk = nt * 16 + m;
      if (colInBlk < BDIM) {             // skip pad columns (avoid clobbering next block)
        int colD = b * BDIM + colInBlk;
        #pragma unroll
        for (int i = 0; i < 8; ++i)
          su.msg[wid][half * 8 + i][colD] = acc[i];
      }
    }
  }
  __syncthreads();

  // row-uniform, lane-coalesced atomic scatter (scalar branch per row)
  for (int r = 0; r < 16; ++r) {
    int d = sDst[wid][r];
    if (d < 0) continue;
    float nrm = sNrm[wid][r];
    for (int c = lane; c < DD; c += 32)
      atomicAdd(&agg[(size_t)d * DD + c], su.msg[wid][r][c] * nrm);
  }
}

// ---------------- WMMA root GEMM + bias + residual + LN + ReLU ----------------
// One workgroup = 64 rows (4 waves x 16). A from padded bf16 Xh/Xl (global),
// B from pre-transposed padded root weights (global, L1-hot). Wave-local epilogue.
__global__ __launch_bounds__(128)
void k_update(const unsigned short* __restrict__ Xh, const unsigned short* __restrict__ Xl,
              const float* __restrict__ xin, const float* __restrict__ agg,
              const unsigned short* __restrict__ RWth, const unsigned short* __restrict__ RWtl,
              const float* __restrict__ rootB, const float* __restrict__ lnS,
              const float* __restrict__ lnB, float* __restrict__ xout, int addRes) {
  __shared__ __align__(16) float sy[4][16][128];
  int tid = threadIdx.x, wid = tid >> 5, lane = tid & 31;
  int m = lane & 15, half = lane >> 4;
  int base = blockIdx.x * 64 + wid * 16;
  int row = base + m;

  BFrag ah[4], al[4];
  #pragma unroll
  for (int kc = 0; kc < 4; ++kc) {
    #pragma unroll
    for (int v = 0; v < 8; ++v) {
      int kb = kc * 32 + half * 8 + ((v < 4) ? 2 * v : 16 + 2 * (v - 4));
      unsigned hv = 0u, lv = 0u;
      if (row < NN) {
        hv = *reinterpret_cast<const unsigned*>(Xh + (size_t)row * 128 + kb);
        lv = *reinterpret_cast<const unsigned*>(Xl + (size_t)row * 128 + kb);
      }
      ah[kc].u32[v] = hv;
      al[kc].u32[v] = lv;
    }
  }

  #pragma unroll
  for (int nt = 0; nt < 8; ++nt) {
    int ncol = nt * 16 + m;
    f32x8 acc = {};
    #pragma unroll
    for (int kc = 0; kc < 4; ++kc) {
      BFrag bh, bl;
      #pragma unroll
      for (int v = 0; v < 8; ++v) {
        int kk = kc * 32 + half * 8 + ((v < 4) ? 2 * v : 16 + 2 * (v - 4));
        bh.u32[v] = *reinterpret_cast<const unsigned*>(RWth + (size_t)ncol * 128 + kk);
        bl.u32[v] = *reinterpret_cast<const unsigned*>(RWtl + (size_t)ncol * 128 + kk);
      }
      acc = __builtin_amdgcn_wmma_f32_16x16x32_bf16(false, ah[kc].v, false, bh.v, (short)0, acc, false, false);
      acc = __builtin_amdgcn_wmma_f32_16x16x32_bf16(false, ah[kc].v, false, bl.v, (short)0, acc, false, false);
      acc = __builtin_amdgcn_wmma_f32_16x16x32_bf16(false, al[kc].v, false, bh.v, (short)0, acc, false, false);
    }
    #pragma unroll
    for (int i = 0; i < 8; ++i) sy[wid][half * 8 + i][ncol] = acc[i];
  }

  // wave-local fused epilogue: +agg +bias (+residual), LayerNorm, ReLU
  for (int r = 0; r < 16; ++r) {
    int rr = base + r;
    if (rr >= NN) continue;
    float vv[4];
    float sum = 0.0f;
    #pragma unroll
    for (int j = 0; j < 4; ++j) {
      int c = lane + 32 * j;
      float t = 0.0f;
      if (c < DD) {
        t = sy[wid][r][c] + agg[(size_t)rr * DD + c] + rootB[c];
        if (addRes) t += xin[(size_t)rr * DD + c];
      }
      vv[j] = t;
      sum += t;
    }
    #pragma unroll
    for (int off = 16; off > 0; off >>= 1) sum += __shfl_xor(sum, off);
    float mu = sum * (1.0f / DD);
    float sq = 0.0f;
    #pragma unroll
    for (int j = 0; j < 4; ++j) {
      int c = lane + 32 * j;
      if (c < DD) { float d = vv[j] - mu; sq += d * d; }
    }
    #pragma unroll
    for (int off = 16; off > 0; off >>= 1) sq += __shfl_xor(sq, off);
    float rs = rsqrtf(sq * (1.0f / DD) + 1e-5f);
    #pragma unroll
    for (int j = 0; j < 4; ++j) {
      int c = lane + 32 * j;
      if (c < DD)
        xout[(size_t)rr * DD + c] = fmaxf((vv[j] - mu) * rs * lnS[c] + lnB[c], 0.0f);
    }
  }
}

// ---------------- host side ----------------
extern "C" void kernel_launch(void* const* d_in, const int* in_sizes, int n_in,
                              void* d_out, int out_size, void* d_ws, size_t ws_size,
                              hipStream_t stream) {
  (void)in_sizes; (void)n_in; (void)out_size; (void)ws_size;
  const int*   ei    = (const int*)d_in[0];
  const int*   etyp  = (const int*)d_in[1];
  const float* ent   = (const float*)d_in[2];
  const float* relW  = (const float*)d_in[3];
  const float* rootW = (const float*)d_in[4];
  const float* rootB = (const float*)d_in[5];
  const float* lnS   = (const float*)d_in[6];
  const float* lnB   = (const float*)d_in[7];
  float* out = (float*)d_out;
  const int* srcp = ei;
  const int* dstp = ei + EE;

  char* ws = (char*)d_ws;
  size_t o = 0;
  auto carve = [&](size_t bytes) -> void* {
    void* p = ws + o;
    o = (o + bytes + 255) & ~(size_t)255;
    return p;
  };
  unsigned* cnt      = (unsigned*)carve((size_t)NN * RR * 4);
  unsigned* rel_cnt  = (unsigned*)carve(RR * 4);
  unsigned* rel_off  = (unsigned*)carve((RR + 1) * 4);
  unsigned* rel_cur  = (unsigned*)carve(RR * 4);
  unsigned* n_tiles  = (unsigned*)carve(4);
  uint2*    tiles    = (uint2*)carve((size_t)MAX_TILES * 8);
  unsigned* eord     = (unsigned*)carve((size_t)EE * 4);
  unsigned short* Wh = (unsigned short*)carve((size_t)LL * RR * NBLK * BDIM * BDIM * 2);
  unsigned short* Wl = (unsigned short*)carve((size_t)LL * RR * NBLK * BDIM * BDIM * 2);
  unsigned short* RWth = (unsigned short*)carve((size_t)LL * 128 * 128 * 2);
  unsigned short* RWtl = (unsigned short*)carve((size_t)LL * 128 * 128 * 2);
  unsigned short* Xh = (unsigned short*)carve((size_t)NN * 128 * 2);
  unsigned short* Xl = (unsigned short*)carve((size_t)NN * 128 * 2);
  float* x1  = (float*)carve((size_t)NN * DD * 4);
  float* agg = (float*)carve((size_t)NN * DD * 4);

  // layer-invariant prep (recomputed each call for determinism)
  k_zero_u32<<<dim3((NN * RR + 255) / 256), 256, 0, stream>>>(cnt, (long)NN * RR);
  k_zero_u32<<<dim3(1), 256, 0, stream>>>(rel_cnt, (long)RR);
  k_count<<<dim3((EE + 255) / 256), 256, 0, stream>>>(dstp, etyp, cnt, rel_cnt);
  k_scan_tiles<<<dim3(1), 1, 0, stream>>>(rel_cnt, rel_off, rel_cur, tiles, n_tiles);
  k_scatter<<<dim3((EE + 255) / 256), 256, 0, stream>>>(etyp, rel_cur, eord);
  const long wn = (long)LL * RR * NBLK * BDIM * BDIM;
  k_convW<<<dim3((unsigned)((wn + 255) / 256)), 256, 0, stream>>>(relW, Wh, Wl, wn);
  k_convRW<<<dim3((LL * 128 * 128 + 255) / 256), 256, 0, stream>>>(rootW, RWth, RWtl);

  const size_t wlayer = (size_t)RR * NBLK * BDIM * BDIM;
  for (int l = 0; l < LL; ++l) {
    const float* xin = (l == 0) ? ent : x1;
    float* xout = (l == LL - 1) ? out : x1;
    k_convX<<<dim3(((long)NN * 128 + 255) / 256), 256, 0, stream>>>(xin, Xh, Xl);
    k_zero_f32<<<dim3((NN * DD + 255) / 256), 256, 0, stream>>>(agg, (long)NN * DD);
    k_msg<<<dim3(MAX_TILES), 128, 0, stream>>>(Xh, Xl, srcp, dstp, eord, cnt, rel_off,
                                               tiles, n_tiles,
                                               Wh + l * wlayer, Wl + l * wlayer, agg);
    k_update<<<dim3((NN + 63) / 64), 128, 0, stream>>>(Xh, Xl, xin, agg,
                                                       RWth + (size_t)l * 128 * 128,
                                                       RWtl + (size_t)l * 128 * 128,
                                                       rootB + (size_t)l * DD,
                                                       lnS + (size_t)l * DD,
                                                       lnB + (size_t)l * DD,
                                                       xout, (l > 0) ? 1 : 0);
  }
}
